// Decoder_45157286150890
// MI455X (gfx1250) — matrix operand order repaired and checked by reading, not compiled, not run on hardware
//
#include <hip/hip_runtime.h>
#include <hip/hip_bf16.h>

typedef __bf16 bf16_t;
typedef __attribute__((ext_vector_type(16))) __bf16 v16bf;
typedef __attribute__((ext_vector_type(8)))  __bf16 v8bf;
typedef __attribute__((ext_vector_type(8)))  float  v8f;

#define CAT16(a, b) __builtin_shufflevector((a), (b), 0,1,2,3,4,5,6,7,8,9,10,11,12,13,14,15)

__device__ __forceinline__ bf16_t f2bf(float f) {
  unsigned u = __builtin_bit_cast(unsigned, f);
  u += 0x7fffu + ((u >> 16) & 1u);            // round-to-nearest-even
  unsigned short s = (unsigned short)(u >> 16);
  return __builtin_bit_cast(bf16_t, s);
}

__device__ __forceinline__ v8f wmma_bf16(v16bf a, v16bf b, v8f c) {
  return __builtin_amdgcn_wmma_f32_16x16x32_bf16(false, a, false, b, (short)0, c, false, false);
}

// A fragment (16x32 bf16, M=l&15, contiguous K run at 16*hi) from bf16 row-major
__device__ __forceinline__ v16bf load_afrag_bf16(const bf16_t* arow_k, int hi) {
  v8bf a1 = *(const v8bf*)(arow_k + 16 * hi);
  v8bf a2 = *(const v8bf*)(arow_k + 16 * hi + 8);
  return CAT16(a1, a2);
}

// B fragment (32x16 bf16, N=l&15, contiguous K run; caller applied 16*hi)
__device__ __forceinline__ v16bf load_bfrag_bf16(const bf16_t* brow_k) {
  v8bf b1 = *(const v8bf*)(brow_k);
  v8bf b2 = *(const v8bf*)(brow_k + 8);
  return CAT16(b1, b2);
}

// A fragment converted on the fly from f32 row-major
__device__ __forceinline__ v16bf load_afrag_f32(const float* row_k, int hi) {
  const float* p1 = row_k + 16 * hi;
  const float* p2 = p1 + 8;
  float4 q0 = *(const float4*)p1;
  float4 q1 = *(const float4*)(p1 + 4);
  float4 q2 = *(const float4*)p2;
  float4 q3 = *(const float4*)(p2 + 4);
  v16bf a;
  a[0] = f2bf(q0.x);  a[1] = f2bf(q0.y);  a[2] = f2bf(q0.z);  a[3] = f2bf(q0.w);
  a[4] = f2bf(q1.x);  a[5] = f2bf(q1.y);  a[6] = f2bf(q1.z);  a[7] = f2bf(q1.w);
  a[8] = f2bf(q2.x);  a[9] = f2bf(q2.y);  a[10] = f2bf(q2.z); a[11] = f2bf(q2.w);
  a[12] = f2bf(q3.x); a[13] = f2bf(q3.y); a[14] = f2bf(q3.z); a[15] = f2bf(q3.w);
  return a;
}

// ---------------------------------------------------------------------------
// Generic WMMA GEMM: C[M,N] = act( (acc ? C : 0) + A[M,K] * B[N,K]^T + bias + bias2 )
// 4 waves/WG, each wave does a 16x64 strip. M multiple of 64, K compile-time.
// GUARD: out-of-range columns get clamped load addresses (row N-1); their
// accumulator columns are garbage but are never stored (WMMA columns are
// independent), so no zero-select is needed in the hot loop and EXEC stays
// all-ones. B fragments are explicitly double-buffered across k-steps.
// ---------------------------------------------------------------------------
template <bool GUARD, int K>
__global__ __launch_bounds__(128) void gemm_bf16_kernel(
    const bf16_t* __restrict__ A, const bf16_t* __restrict__ Bm,
    const float* __restrict__ bias, const float* __restrict__ bias2,
    float* __restrict__ C, bf16_t* __restrict__ Cb,
    int M, int N, int accFlag, int reluFlag) {
  const int tid = threadIdx.x;
  const int wave = tid >> 5, l = tid & 31;
  const int lo = l & 15, hi = l >> 4;
  const int row0 = blockIdx.x * 64 + wave * 16;
  const int colbase = blockIdx.y * 64;

  v8f acc[4];
#pragma unroll
  for (int i = 0; i < 4; ++i) {
    int n = colbase + i * 16 + lo;
    if (accFlag && (!GUARD || n < N)) {
#pragma unroll
      for (int j = 0; j < 8; ++j) acc[i][j] = C[(size_t)(row0 + j + 8 * hi) * N + n];
    } else {
#pragma unroll
      for (int j = 0; j < 8; ++j) acc[i][j] = 0.0f;
    }
  }

  const bf16_t* arow = A + (size_t)(row0 + lo) * K;
  // per-column-block B row pointers; clamped when GUARD (OOB columns discarded
  // in the epilogue). When !GUARD these collapse to base + immediate offsets.
  const bf16_t* br0;
  const bf16_t* br1;
  const bf16_t* br2;
  const bf16_t* br3;
  if (!GUARD) {
    const bf16_t* bbase = Bm + (size_t)(colbase + lo) * K + 16 * hi;
    br0 = bbase + 0 * 16 * K;
    br1 = bbase + 1 * 16 * K;
    br2 = bbase + 2 * 16 * K;
    br3 = bbase + 3 * 16 * K;
  } else {
    int n0 = colbase + 0 * 16 + lo, n1 = colbase + 1 * 16 + lo;
    int n2 = colbase + 2 * 16 + lo, n3 = colbase + 3 * 16 + lo;
    br0 = Bm + (size_t)(n0 < N ? n0 : N - 1) * K + 16 * hi;
    br1 = Bm + (size_t)(n1 < N ? n1 : N - 1) * K + 16 * hi;
    br2 = Bm + (size_t)(n2 < N ? n2 : N - 1) * K + 16 * hi;
    br3 = Bm + (size_t)(n3 < N ? n3 : N - 1) * K + 16 * hi;
  }

  // software pipeline: B fragments double-buffered across k-steps
  v16bf b0 = load_bfrag_bf16(br0);
  v16bf b1 = load_bfrag_bf16(br1);
  v16bf b2 = load_bfrag_bf16(br2);
  v16bf b3 = load_bfrag_bf16(br3);
#pragma unroll
  for (int k = 0; k < K; k += 32) {
    v16bf n0, n1, n2, n3;
    if (k + 32 < K) {
      n0 = load_bfrag_bf16(br0 + k + 32);
      n1 = load_bfrag_bf16(br1 + k + 32);
      n2 = load_bfrag_bf16(br2 + k + 32);
      n3 = load_bfrag_bf16(br3 + k + 32);
    }
    v16bf a = load_afrag_bf16(arow + k, hi);
    acc[0] = wmma_bf16(a, b0, acc[0]);
    acc[1] = wmma_bf16(a, b1, acc[1]);
    acc[2] = wmma_bf16(a, b2, acc[2]);
    acc[3] = wmma_bf16(a, b3, acc[3]);
    if (k + 32 < K) { b0 = n0; b1 = n1; b2 = n2; b3 = n3; }
  }

#pragma unroll
  for (int i = 0; i < 4; ++i) {
    int n = colbase + i * 16 + lo;
    if (GUARD && n >= N) continue;
    float bv = (bias ? bias[n] : 0.0f) + (bias2 ? bias2[n] : 0.0f);
#pragma unroll
    for (int j = 0; j < 8; ++j) {
      float v = acc[i][j] + bv;
      if (reluFlag) v = fmaxf(v, 0.0f);
      int r = row0 + j + 8 * hi;
      C[(size_t)r * N + n] = v;
      if (Cb) Cb[(size_t)r * N + n] = f2bf(v);
    }
  }
}

// ---------------------------------------------------------------------------
// Fused additive-attention score:
//   score[b,s] = sum_d tanh( dec[b,d] + sum_e enc[s,b,e]*We[d,e] ) * v[d]
// enc f32 [S,B,E] (== row-major [S*B,E]); dec f32 [B,512] (bias folded in);
// We bf16 [512,E]. 32 rows (one s, 32 b) per 256-thread WG; 8 waves, each wave
// owns 64 d-columns x 2 row-tiles so B fragments are reused across row-tiles.
// E compile-time -> single B base + immediate offsets; B fragments explicitly
// double-buffered across k-steps. Never materializes the [S,B,512] tensor.
// ---------------------------------------------------------------------------
template <int E>
__global__ __launch_bounds__(256) void attn_score_kernel(
    const float* __restrict__ enc, const float* __restrict__ dec,
    const bf16_t* __restrict__ We, const float* __restrict__ v,
    float* __restrict__ score, int S, int B) {
  const int tid = threadIdx.x;
  const int wave = tid >> 5, l = tid & 31;
  const int lo = l & 15, hi = l >> 4;
  const int r0 = blockIdx.x * 32;
  const int s = r0 / B, b0 = r0 % B;
  const int d0 = wave * 64;

  v8f acc[2][4];
#pragma unroll
  for (int rt = 0; rt < 2; ++rt)
#pragma unroll
    for (int i = 0; i < 4; ++i) {
      int d = d0 + i * 16 + lo;
#pragma unroll
      for (int j = 0; j < 8; ++j)
        acc[rt][i][j] = dec[(b0 + rt * 16 + j + 8 * hi) * 512 + d];
    }

  const bf16_t* bbase = We + (size_t)(d0 + lo) * E + 16 * hi;
  const float* erow0 = enc + (size_t)(r0 + lo) * E;
  const float* erow1 = enc + (size_t)(r0 + 16 + lo) * E;

  // software pipeline: B (weight) fragments double-buffered across k-steps
  v16bf w0 = load_bfrag_bf16(bbase + 0 * 16 * E);
  v16bf w1 = load_bfrag_bf16(bbase + 1 * 16 * E);
  v16bf w2 = load_bfrag_bf16(bbase + 2 * 16 * E);
  v16bf w3 = load_bfrag_bf16(bbase + 3 * 16 * E);
#pragma unroll
  for (int k = 0; k < E; k += 32) {
    v16bf x0, x1, x2, x3;
    if (k + 32 < E) {
      x0 = load_bfrag_bf16(bbase + 0 * 16 * E + k + 32);
      x1 = load_bfrag_bf16(bbase + 1 * 16 * E + k + 32);
      x2 = load_bfrag_bf16(bbase + 2 * 16 * E + k + 32);
      x3 = load_bfrag_bf16(bbase + 3 * 16 * E + k + 32);
      __builtin_prefetch(erow0 + k + 64, 0, 3);  // stream-ahead on enc
      __builtin_prefetch(erow1 + k + 64, 0, 3);
    }
    v16bf a0 = load_afrag_f32(erow0 + k, hi);
    acc[0][0] = wmma_bf16(a0, w0, acc[0][0]);
    acc[0][1] = wmma_bf16(a0, w1, acc[0][1]);
    acc[0][2] = wmma_bf16(a0, w2, acc[0][2]);
    acc[0][3] = wmma_bf16(a0, w3, acc[0][3]);
    v16bf a1 = load_afrag_f32(erow1 + k, hi);
    acc[1][0] = wmma_bf16(a1, w0, acc[1][0]);
    acc[1][1] = wmma_bf16(a1, w1, acc[1][1]);
    acc[1][2] = wmma_bf16(a1, w2, acc[1][2]);
    acc[1][3] = wmma_bf16(a1, w3, acc[1][3]);
    if (k + 32 < E) { w0 = x0; w1 = x1; w2 = x2; w3 = x3; }
  }

  __shared__ float sred[8][32];
#pragma unroll
  for (int rt = 0; rt < 2; ++rt) {
    float ps[8];
#pragma unroll
    for (int j = 0; j < 8; ++j) ps[j] = 0.0f;
#pragma unroll
    for (int i = 0; i < 4; ++i) {
      float vv = v[d0 + i * 16 + lo];
#pragma unroll
      for (int j = 0; j < 8; ++j) ps[j] += tanhf(acc[rt][i][j]) * vv;
    }
    // reduce over the 16 column-lanes of each half (rows differ between halves)
#pragma unroll
    for (int j = 0; j < 8; ++j) {
      float t = ps[j];
      t += __shfl_xor(t, 1, 32);
      t += __shfl_xor(t, 2, 32);
      t += __shfl_xor(t, 4, 32);
      t += __shfl_xor(t, 8, 32);
      if (lo == 0) sred[wave][rt * 16 + j + 8 * hi] = t;
    }
  }
  __syncthreads();
  if (tid < 32) {
    float sum = 0.0f;
#pragma unroll
    for (int w2i = 0; w2i < 8; ++w2i) sum += sred[w2i][tid];
    score[(size_t)(b0 + tid) * S + s] = sum;
  }
}

// softmax over S, in place; one block per b
__global__ __launch_bounds__(128) void softmax_kernel(float* __restrict__ score, int S) {
  const int b = blockIdx.x, t = threadIdx.x;
  float* row = score + (size_t)b * S;
  __shared__ float red[128];
  float m = -__builtin_inff();
  for (int s = t; s < S; s += 128) m = fmaxf(m, row[s]);
  red[t] = m; __syncthreads();
  for (int o = 64; o; o >>= 1) { if (t < o) red[t] = fmaxf(red[t], red[t + o]); __syncthreads(); }
  m = red[0]; __syncthreads();
  float sum = 0.0f;
  for (int s = t; s < S; s += 128) { float e = expf(row[s] - m); row[s] = e; sum += e; }
  red[t] = sum; __syncthreads();
  for (int o = 64; o; o >>= 1) { if (t < o) red[t] += red[t + o]; __syncthreads(); }
  float inv = 1.0f / red[0];
  for (int s = t; s < S; s += 128) row[s] *= inv;
}

// ctx[b,e] = sum_s w[b,s] * enc[s,b,e];  grid (B, E/64), 64 threads
__global__ void ctx_kernel(const float* __restrict__ w, const float* __restrict__ enc,
                           float* __restrict__ out, bf16_t* __restrict__ outb,
                           int S, int B, int E) {
  const int b = blockIdx.x;
  const int e = blockIdx.y * 64 + threadIdx.x;
  const float* wrow = w + (size_t)b * S;
  const float* ep = enc + (size_t)b * E + e;
  const size_t stride = (size_t)B * E;
  float sum = 0.0f;
  for (int s = 0; s < S; ++s) sum += wrow[s] * ep[(size_t)s * stride];
  out[(size_t)b * E + e] = sum;
  if (outb) outb[(size_t)b * E + e] = f2bf(sum);
}

// LSTM elementwise from pre-activation gates G[B,4D] (i,f,g,o)
__global__ void lstm_elem_kernel(const float* __restrict__ G, const float* __restrict__ cprev,
                                 float* __restrict__ hout, float* __restrict__ cout,
                                 bf16_t* __restrict__ hb, int B, int D) {
  int idx = blockIdx.x * blockDim.x + threadIdx.x;
  if (idx >= B * D) return;
  int b = idx / D, d = idx - b * D;
  const float* g = G + (size_t)b * 4 * D;
  float gi = g[d], gf = g[D + d], gg = g[2 * D + d], go = g[3 * D + d];
  float si = 1.0f / (1.0f + expf(-gi));
  float sf = 1.0f / (1.0f + expf(-gf));
  float so = 1.0f / (1.0f + expf(-go));
  float c2 = sf * cprev[idx] + si * tanhf(gg);
  float h2 = so * tanhf(c2);
  hout[idx] = h2; cout[idx] = c2; hb[idx] = f2bf(h2);
}

// x[b,0]=ts, x[b,1..64]=embed[cat[b]], x[b,65..127]=0  (bf16, padded K=128)
__global__ void embed_concat_kernel(const float* __restrict__ p_ts, const int* __restrict__ p_cat,
                                    const float* __restrict__ emb, bf16_t* __restrict__ xb, int B) {
  int idx = blockIdx.x * blockDim.x + threadIdx.x;
  if (idx >= B * 128) return;
  int b = idx >> 7, c = idx & 127;
  float v;
  if (c == 0)       v = p_ts[b];
  else if (c <= 64) v = emb[(size_t)p_cat[b] * 64 + (c - 1)];
  else              v = 0.0f;
  xb[idx] = f2bf(v);
}

// dst_bf16[r, 0..dcols) = src_f32[r, coff .. ] (zero-padded past scols)
__global__ void convert_sub_kernel(const float* __restrict__ src, bf16_t* __restrict__ dst,
                                   int rows, int dcols, int scols, int sld, int coff) {
  int idx = blockIdx.x * blockDim.x + threadIdx.x;
  if (idx >= rows * dcols) return;
  int r = idx / dcols, c = idx - r * dcols;
  float v = (c < scols) ? src[(size_t)r * sld + coff + c] : 0.0f;
  dst[idx] = f2bf(v);
}

__global__ __launch_bounds__(256) void log_softmax_kernel(const float* __restrict__ logits,
                                                          float* __restrict__ out, int V) {
  const int b = blockIdx.x, t = threadIdx.x;
  const float* row = logits + (size_t)b * V;
  __shared__ float red[256];
  float m = -__builtin_inff();
  for (int i = t; i < V; i += 256) m = fmaxf(m, row[i]);
  red[t] = m; __syncthreads();
  for (int o = 128; o; o >>= 1) { if (t < o) red[t] = fmaxf(red[t], red[t + o]); __syncthreads(); }
  m = red[0]; __syncthreads();
  float sum = 0.0f;
  for (int i = t; i < V; i += 256) sum += expf(row[i] - m);
  red[t] = sum; __syncthreads();
  for (int o = 128; o; o >>= 1) { if (t < o) red[t] += red[t + o]; __syncthreads(); }
  float lse = m + logf(red[0]);
  for (int i = t; i < V; i += 256) out[(size_t)b * V + i] = row[i] - lse;
}

__global__ void ts_kernel(const float* __restrict__ h, const float* __restrict__ tgW,
                          const float* __restrict__ tgb, float* __restrict__ out, int B, int D) {
  int b = blockIdx.x * blockDim.x + threadIdx.x;
  if (b >= B) return;
  float s = tgb[0];
  for (int d = 0; d < D; ++d) s += h[(size_t)b * D + d] * tgW[d];
  out[b] = fmaxf(s, 0.0f);
}

// ---------------------------------------------------------------------------
extern "C" void kernel_launch(void* const* d_in, const int* in_sizes, int n_in,
                              void* d_out, int out_size, void* d_ws, size_t ws_size,
                              hipStream_t stream) {
  (void)in_sizes; (void)n_in; (void)out_size; (void)ws_size;
  const int B = 256, D = 512, S = 512, PD = 512, OD = 256, ID = 1024, V = 1000, G4 = 2048;

  const float* p_ts   = (const float*)d_in[0];
  const int*   p_cat  = (const int*)  d_in[1];
  const float* p_hn   = (const float*)d_in[2];
  const float* p_hc   = (const float*)d_in[3];
  const float* p_enc  = (const float*)d_in[4];
  const float* a_enc  = (const float*)d_in[5];
  const float* i_enc  = (const float*)d_in[6];
  const float* embed  = (const float*)d_in[7];
  const float* W_ih0  = (const float*)d_in[8];  const float* W_hh0 = (const float*)d_in[9];
  const float* b_ih0  = (const float*)d_in[10]; const float* b_hh0 = (const float*)d_in[11];
  const float* W_ih1  = (const float*)d_in[12]; const float* W_hh1 = (const float*)d_in[13];
  const float* b_ih1  = (const float*)d_in[14]; const float* b_hh1 = (const float*)d_in[15];
  const float* pW = (const float*)d_in[16]; const float* pb = (const float*)d_in[17]; const float* pv = (const float*)d_in[18];
  const float* aW = (const float*)d_in[19]; const float* ab = (const float*)d_in[20]; const float* av = (const float*)d_in[21];
  const float* iW = (const float*)d_in[22]; const float* ib = (const float*)d_in[23]; const float* iv = (const float*)d_in[24];
  const float* cW = (const float*)d_in[25]; const float* cb = (const float*)d_in[26]; const float* cv = (const float*)d_in[27];
  const float* p2o_W = (const float*)d_in[28]; const float* p2o_b = (const float*)d_in[29];
  const float* o2p_W = (const float*)d_in[30]; const float* o2p_b = (const float*)d_in[31];
  const float* out1_W = (const float*)d_in[32]; const float* out1_b = (const float*)d_in[33];
  const float* out2_W = (const float*)d_in[34]; const float* out2_b = (const float*)d_in[35];
  const float* mg_W = (const float*)d_in[36]; const float* mg_b = (const float*)d_in[37];
  const float* tg_W = (const float*)d_in[38]; const float* tg_b = (const float*)d_in[39];

  float* out_f = (float*)d_out;
  float* out_ts = out_f;                     // [B]
  float* out_cat = out_f + B;                // [B,V]
  float* out_hn = out_f + B + B * V;         // [2,B,D]
  float* out_hc = out_hn + 2 * B * D;        // [2,B,D]

  // ---- workspace bump allocator ----
  char* wsp = (char*)d_ws;
  size_t off = 0;
  auto alloc = [&](size_t bytes) -> void* {
    void* p = wsp + off;
    off = (off + bytes + 255) & ~(size_t)255;
    return p;
  };
  auto abf = [&](size_t n) { return (bf16_t*)alloc(n * 2); };
  auto af  = [&](size_t n) { return (float*)alloc(n * 4); };

  bf16_t* xb     = abf(B * 128);
  bf16_t* h0pb   = abf(B * D);
  bf16_t* h1pb   = abf(B * D);
  bf16_t* Wih0b  = abf(G4 * 128);
  bf16_t* Whh0b  = abf(G4 * D);
  bf16_t* Wih1b  = abf(G4 * D);
  bf16_t* Whh1b  = abf(G4 * D);
  bf16_t* pWhb   = abf(D * D);
  bf16_t* pWeb   = abf(D * PD);
  bf16_t* aWhb   = abf(D * D);
  bf16_t* aWeb   = abf(D * OD);
  bf16_t* iWhb   = abf(D * D);
  bf16_t* iWeb   = abf(D * OD);
  bf16_t* cWhb   = abf(D * D);
  bf16_t* cWeb   = abf(D * OD);
  bf16_t* p2oWb  = abf(OD * D);
  bf16_t* o2pWb  = abf(D * OD);
  bf16_t* out1Wa = abf(ID * D);
  bf16_t* out1Wc = abf(ID * D);
  bf16_t* out2Wb = abf(D * ID);
  bf16_t* mgWb   = abf(V * D);
  bf16_t* h0b    = abf(B * D);
  bf16_t* h1b    = abf(B * D);
  bf16_t* ctxpb  = abf(B * PD);
  bf16_t* ctx2b  = abf(B * OD);
  bf16_t* contextb = abf(B * D);
  bf16_t* hmidb  = abf(B * ID);
  bf16_t* h2b    = abf(B * D);

  float* G0    = af(B * G4);
  float* G1    = af(B * G4);
  float* decp  = af(B * D);
  float* deca  = af(B * D);
  float* deci  = af(B * D);
  float* decc  = af(B * D);
  float* sc_p  = af(B * S);
  float* sc_a  = af(B * S);
  float* sc_i  = af(B * S);
  float* sc_c  = af(B * 3);
  float* ctxp  = af(B * PD);
  float* comb  = af(3 * B * OD);   // [3,B,OD]
  float* ctx2  = af(B * OD);
  float* ctxD  = af(B * D);        // "context" after o2p
  float* hmid  = af(B * ID);
  float* h2    = af(B * D);
  float* logit = af(B * V);

  auto conv = [&](const float* src, bf16_t* dst, int rows, int dcols, int scols, int sld, int coff) {
    int n = rows * dcols;
    convert_sub_kernel<<<(n + 255) / 256, 256, 0, stream>>>(src, dst, rows, dcols, scols, sld, coff);
  };
  auto gemm = [&](const bf16_t* A, const bf16_t* Bm, const float* bias, const float* bias2,
                  float* C, bf16_t* Cb, int M, int N, int K, int acc, int relu) {
    dim3 g(M / 64, (N + 63) / 64);
    bool guard = (N % 64) != 0;
    if (K == 128) {
      if (guard) gemm_bf16_kernel<true, 128><<<g, 128, 0, stream>>>(A, Bm, bias, bias2, C, Cb, M, N, acc, relu);
      else       gemm_bf16_kernel<false, 128><<<g, 128, 0, stream>>>(A, Bm, bias, bias2, C, Cb, M, N, acc, relu);
    } else if (K == 256) {
      if (guard) gemm_bf16_kernel<true, 256><<<g, 128, 0, stream>>>(A, Bm, bias, bias2, C, Cb, M, N, acc, relu);
      else       gemm_bf16_kernel<false, 256><<<g, 128, 0, stream>>>(A, Bm, bias, bias2, C, Cb, M, N, acc, relu);
    } else if (K == 512) {
      if (guard) gemm_bf16_kernel<true, 512><<<g, 128, 0, stream>>>(A, Bm, bias, bias2, C, Cb, M, N, acc, relu);
      else       gemm_bf16_kernel<false, 512><<<g, 128, 0, stream>>>(A, Bm, bias, bias2, C, Cb, M, N, acc, relu);
    } else { // 1024
      if (guard) gemm_bf16_kernel<true, 1024><<<g, 128, 0, stream>>>(A, Bm, bias, bias2, C, Cb, M, N, acc, relu);
      else       gemm_bf16_kernel<false, 1024><<<g, 128, 0, stream>>>(A, Bm, bias, bias2, C, Cb, M, N, acc, relu);
    }
  };

  // ---- stage 0: input prep / weight conversion ----
  embed_concat_kernel<<<(B * 128 + 255) / 256, 256, 0, stream>>>(p_ts, p_cat, embed, xb, B);
  conv(p_hn,          h0pb, B, D, D, D, 0);
  conv(p_hn + B * D,  h1pb, B, D, D, D, 0);
  conv(W_ih0, Wih0b, G4, 128, 65, 65, 0);        // pad K 65 -> 128
  conv(W_hh0, Whh0b, G4, D, D, D, 0);
  conv(W_ih1, Wih1b, G4, D, D, D, 0);
  conv(W_hh1, Whh1b, G4, D, D, D, 0);
  conv(pW, pWhb, D, D, D, PD + D, 0);
  conv(pW, pWeb, D, PD, PD, PD + D, D);
  conv(aW, aWhb, D, D, D, OD + D, 0);
  conv(aW, aWeb, D, OD, OD, OD + D, D);
  conv(iW, iWhb, D, D, D, OD + D, 0);
  conv(iW, iWeb, D, OD, OD, OD + D, D);
  conv(cW, cWhb, D, D, D, OD + D, 0);
  conv(cW, cWeb, D, OD, OD, OD + D, D);
  conv(p2o_W, p2oWb, OD, D, D, D, 0);
  conv(o2p_W, o2pWb, D, OD, OD, OD, 0);
  conv(out1_W, out1Wa, ID, D, D, 2 * D, 0);
  conv(out1_W, out1Wc, ID, D, D, 2 * D, D);
  conv(out2_W, out2Wb, D, ID, ID, ID, 0);
  conv(mg_W, mgWb, V, D, D, D, 0);

  // ---- stage 1: 2-layer LSTM ----
  gemm(xb,   Wih0b, b_ih0, b_hh0, G0, nullptr, B, G4, 128, 0, 0);
  gemm(h0pb, Whh0b, nullptr, nullptr, G0, nullptr, B, G4, D, 1, 0);
  lstm_elem_kernel<<<(B * D + 255) / 256, 256, 0, stream>>>(G0, p_hc, out_hn, out_hc, h0b, B, D);
  gemm(h0b,  Wih1b, b_ih1, b_hh1, G1, nullptr, B, G4, D, 0, 0);
  gemm(h1pb, Whh1b, nullptr, nullptr, G1, nullptr, B, G4, D, 1, 0);
  lstm_elem_kernel<<<(B * D + 255) / 256, 256, 0, stream>>>(G1, p_hc + B * D, out_hn + B * D,
                                                            out_hc + B * D, h1b, B, D);

  // ---- stage 2: decoder-side attention projections (bias folded in) ----
  gemm(h1b, pWhb, pb, nullptr, decp, nullptr, B, D, D, 0, 0);
  gemm(h1b, aWhb, ab, nullptr, deca, nullptr, B, D, D, 0, 0);
  gemm(h1b, iWhb, ib, nullptr, deci, nullptr, B, D, D, 0, 0);
  gemm(h1b, cWhb, cb, nullptr, decc, nullptr, B, D, D, 0, 0);

  // ---- stage 3: fused attention scores (big WMMA GEMMs, tanh, dot-v) ----
  attn_score_kernel<512><<<S * B / 32, 256, 0, stream>>>(p_enc, decp, pWeb, pv, sc_p, S, B);
  attn_score_kernel<256><<<S * B / 32, 256, 0, stream>>>(a_enc, deca, aWeb, av, sc_a, S, B);
  attn_score_kernel<256><<<S * B / 32, 256, 0, stream>>>(i_enc, deci, iWeb, iv, sc_i, S, B);
  softmax_kernel<<<B, 128, 0, stream>>>(sc_p, S);
  softmax_kernel<<<B, 128, 0, stream>>>(sc_a, S);
  softmax_kernel<<<B, 128, 0, stream>>>(sc_i, S);

  // ---- stage 4: contexts ----
  ctx_kernel<<<dim3(B, PD / 64), 64, 0, stream>>>(sc_p, p_enc, ctxp, ctxpb, S, B, PD);
  ctx_kernel<<<dim3(B, OD / 64), 64, 0, stream>>>(sc_a, a_enc, comb + 1 * B * OD, nullptr, S, B, OD);
  ctx_kernel<<<dim3(B, OD / 64), 64, 0, stream>>>(sc_i, i_enc, comb + 2 * B * OD, nullptr, S, B, OD);
  // combined[0] = relu(p_ctx @ p2o_W.T + p2o_b)
  gemm(ctxpb, p2oWb, p2o_b, nullptr, comb, nullptr, B, OD, D, 0, 1);

  // ---- stage 5: combined attention over S=3 ----
  attn_score_kernel<256><<<3 * B / 32, 256, 0, stream>>>(comb, decc, cWeb, cv, sc_c, 3, B);
  softmax_kernel<<<B, 128, 0, stream>>>(sc_c, 3);
  ctx_kernel<<<dim3(B, OD / 64), 64, 0, stream>>>(sc_c, comb, ctx2, ctx2b, 3, B, OD);

  // ---- stage 6: output head ----
  gemm(ctx2b, o2pWb, o2p_b, nullptr, ctxD, contextb, B, D, OD, 0, 1);       // context
  gemm(contextb, out1Wa, out1_b, nullptr, hmid, nullptr, B, ID, D, 0, 0);   // context part
  gemm(h1b, out1Wc, nullptr, nullptr, hmid, hmidb, B, ID, D, 1, 1);         // + p_output, relu
  gemm(hmidb, out2Wb, out2_b, nullptr, h2, h2b, B, D, ID, 0, 1);            // relu
  gemm(h2b, mgWb, mg_b, nullptr, logit, nullptr, B, V, D, 0, 0);            // logits

  ts_kernel<<<1, 256, 0, stream>>>(h2, tg_W, tg_b, out_ts, B, D);
  log_softmax_kernel<<<B, 256, 0, stream>>>(logit, out_cat, V);
}
